// S4DKernel_67568425501006
// MI455X (gfx1250) — matrix operand order repaired and checked by reading, not compiled, not run on hardware
//
#include <hip/hip_runtime.h>
#include <hip/hip_bf16.h>

typedef __attribute__((ext_vector_type(16))) __bf16          v16bf;
typedef __attribute__((ext_vector_type(8)))  float           v8f;
typedef __attribute__((ext_vector_type(4)))  float           v4f;
typedef __attribute__((ext_vector_type(16))) unsigned short  ushort16_t;
typedef __attribute__((ext_vector_type(4)))  unsigned short  ushort4_t;
typedef __attribute__((ext_vector_type(4)))  unsigned int    uint32x4;
typedef __attribute__((ext_vector_type(8)))  int             int32x8;
typedef __attribute__((ext_vector_type(4)))  int             int32x4;

#define DMODEL  512
#define DSTATE  64
#define TLEN    2048
#define NBATCH  8
#define USTRIDE 2056                // LDS u row stride in bf16 elems (mult of 8)
#define KOFF    32                  // zero-pad of Kext
#define KSZ     (KOFF + TLEN)       // 2080 elements
#define KUNITS  ((KSZ * 2) / 8)     // 520 x 8-byte units for TDM

__device__ __forceinline__ unsigned short f32_to_bf16(float f) {
    unsigned int x = __builtin_bit_cast(unsigned int, f);
    unsigned int r = x + 0x7FFFu + ((x >> 16) & 1u);   // round-nearest-even
    if ((x & 0x7F800000u) == 0x7F800000u) r = x;       // inf/nan passthrough
    return (unsigned short)(r >> 16);
}

// ---------------------------------------------------------------------------
// Phase A: K[d,t] = sum_n (C*B*dt) * exp(a_n t) * cos(b_n t),  K[d,0] += D[d].
// Emit REVERSED bf16 extended kernel: KR[d,j] = Kext[2079-j],
// Kext[i] = (i>=32) ? K[i-32] : 0. Reversal makes phase-B A-tile loads
// contiguous-ascending (pure ds_load_b128, no half-word shuffles).
// ---------------------------------------------------------------------------
__global__ __launch_bounds__(256) void s4d_kgen(
    const float* __restrict__ Alog, const float* __restrict__ Aang,
    const float* __restrict__ Bp,   const float* __restrict__ Cp,
    const float* __restrict__ Dp,   const float* __restrict__ logdt,
    unsigned short* __restrict__ KRws)
{
    const int d   = blockIdx.x;
    const int tid = threadIdx.x;
    __shared__ float a_s[DSTATE], b_s[DSTATE], w_s[DSTATE];

    const float dt = __expf(logdt[d]);
    if (tid < DSTATE) {
        const int   idx = d * DSTATE + tid;
        const float mag = __expf(Alog[idx]);
        const float ang = Aang[idx];
        const float reA = -mag * __cosf(ang);       // A = -exp(Alog)*e^{i ang}
        const float imA = -mag * __sinf(ang);
        float bt = imA * dt;                        // wrap to principal branch
        bt -= 6.28318530718f * rintf(bt * 0.15915494309f);
        a_s[tid] = reA * dt;
        b_s[tid] = bt;
        w_s[tid] = Cp[idx] * Bp[idx] * dt;          // CB is real
    }
    __syncthreads();

    unsigned short* KR = KRws + (size_t)d * KSZ;
    for (int t = tid; t < TLEN; t += 256) {
        const float tf = (float)t;
        float acc = 0.f;
        #pragma unroll 4
        for (int n = 0; n < DSTATE; ++n)
            acc += w_s[n] * __expf(a_s[n] * tf) * __cosf(b_s[n] * tf);
        if (t == 0) acc += Dp[d];                   // fold D*u into K[0]
        KR[2047 - t] = f32_to_bf16(acc);            // reversed; j = 2079-(32+t)
    }
    if (tid < KOFF) KR[TLEN + tid] = 0;             // reversed zero-pad tail
}

// ---------------------------------------------------------------------------
// Phase B helper: one 16x16x32 bf16 WMMA step.
// A[i,k] = Kext[16m + i - k]  ->  one ascending run of 16 KR elems at rbase.
// B[k,n] = u_bf16[batch n][16J + k], contiguous 16 per lane.
// ---------------------------------------------------------------------------
__device__ __forceinline__ v8f tile_mma(const unsigned short* kr,
                                        const unsigned short* ub,
                                        int rbase, int uoff, v8f acc) {
    ushort16_t av, bv;
    #pragma unroll
    for (int e = 0; e < 8; ++e) {
        av[e]     = kr[rbase + e];
        av[e + 8] = kr[rbase + 8 + e];
    }
    #pragma unroll
    for (int e = 0; e < 16; ++e) bv[e] = ub[uoff + e];
    return __builtin_amdgcn_wmma_f32_16x16x32_bf16(
               false, __builtin_bit_cast(v16bf, av),
               false, __builtin_bit_cast(v16bf, bv),
               (short)0, acc, false, false);
}

// ---------------------------------------------------------------------------
// Phase B: per d, y[:, d, :] = LowerToeplitz(K[d]) @ u[:, d, :]
// One block per d, 8 waves; wave w owns I = w, w+8, ..., w+120.
// KR staged LDS<-global by the Tensor Data Mover (D# per ISA §8.3-8.6).
// ---------------------------------------------------------------------------
__global__ __launch_bounds__(256) void s4d_conv(
    const float* __restrict__ u, const unsigned short* __restrict__ KRws,
    float* __restrict__ y)
{
    const int d   = blockIdx.x;
    const int tid = threadIdx.x;
    // Wave id is uniform per wave: pin it to an SGPR so the J loop runs on
    // scalar compare/branch instead of EXEC masking.
    const int wv  = __builtin_amdgcn_readfirstlane(tid >> 5);

    __shared__ __align__(16) unsigned short u_lds[NBATCH * USTRIDE];
    __shared__ __align__(16) unsigned short kr_lds[KSZ];

    // Stage u[0:8, d, :] as bf16 (NT 128-bit loads -> ds_store_b64)
    for (int c = tid; c < (NBATCH * TLEN / 4); c += 256) {
        const int b  = c >> 9;
        const int t4 = (c & 511) << 2;
        const v4f v = __builtin_nontemporal_load(
            (const v4f*)(u + ((long)(b * DMODEL + d)) * TLEN + t4));
        ushort4_t p;
        p[0] = f32_to_bf16(v[0]); p[1] = f32_to_bf16(v[1]);
        p[2] = f32_to_bf16(v[2]); p[3] = f32_to_bf16(v[3]);
        *(ushort4_t*)(&u_lds[b * USTRIDE + t4]) = p;
    }

#if defined(__has_builtin) && __has_builtin(__builtin_amdgcn_tensor_load_to_lds)
    // TDM: one descriptor, 520 x 8B units, 1 row; issued by wave 0 only
    // (EXEC is ignored by tensor ops; gate at wave granularity).
    if (wv == 0) {
        const unsigned long long ga =
            (unsigned long long)(size_t)(KRws + (size_t)d * KSZ);
        const unsigned int lds_off = (unsigned int)(size_t)(void*)kr_lds;
        uint32x4 g0;
        g0[0] = 1u;                                     // count=1, user D#
        g0[1] = lds_off;                                // lds_addr (bytes)
        g0[2] = (unsigned int)(ga & 0xFFFFFFFFu);       // global_addr[31:0]
        g0[3] = (unsigned int)((ga >> 32) & 0x1FFFFFFu) // global_addr[56:32]
              | (2u << 30);                             // type=2 ("image")
        int32x8 g1;
        g1[0] = (3 << 16);                              // data_size=8B, mask=0
        g1[1] = (int)((unsigned)KUNITS << 16);          // tensor_dim0=520
        g1[2] = (1 << 16);                              // tensor_dim1=1
        g1[3] = (int)((unsigned)KUNITS << 16);          // tile_dim0=520
        g1[4] = 1;                                      // tile_dim1=1
        g1[5] = KUNITS;                                 // tensor_dim0_stride
        g1[6] = (int)((unsigned)KUNITS << 16);          // tensor_dim1_stride
        g1[7] = 0;
        const int32x4 gz4 = {0, 0, 0, 0};
        const int32x8 gz8 = {0, 0, 0, 0, 0, 0, 0, 0};
        __builtin_amdgcn_tensor_load_to_lds(g0, g1, gz4, gz4, gz8, 0);
        __builtin_amdgcn_s_wait_tensorcnt(0);
    }
#else
    for (int t = tid; t < KSZ; t += 256)
        kr_lds[t] = KRws[(size_t)d * KSZ + t];
#endif
    __syncthreads();

    const int lane = tid & 31;
    const int irow = lane & 15;          // A: row M / B,D: column N
    const int hi   = lane >> 4;
    const int kbA  = hi << 4;            // A layout: K = kbA + e (contiguous)
    const int kbB  = hi << 4;            // B layout: K = kbB + e (contiguous)
    const bool act = irow < NBATCH;      // only 8 real batch columns
    const int  bb  = irow & 7;           // cols 8-15 duplicate (never stored)
    const int  rb0 = 2047 - irow + kbA;  // KR base at m=0
    const int  ub0 = bb * USTRIDE + kbB;

    for (int idx = 0; idx < 16; ++idx) {
        const int I = wv + (idx << 3);   // scalar: output row-block
        v8f acc0, acc1;
        #pragma unroll
        for (int q = 0; q < 8; ++q) { acc0[q] = 0.f; acc1[q] = 0.f; }

        // Two independent WMMA chains over J pairs (m = I - J, J even);
        // I and J are SGPRs -> scalar loop control around the DS+WMMA body.
        int J = 0;
        for (; J + 2 <= I; J += 4) {
            acc0 = tile_mma(kr_lds, u_lds, rb0 - ((I - J) << 4),
                            ub0 + (J << 4), acc0);
            acc1 = tile_mma(kr_lds, u_lds, rb0 - ((I - J - 2) << 4),
                            ub0 + ((J + 2) << 4), acc1);
        }
        if (J <= I)
            acc0 = tile_mma(kr_lds, u_lds, rb0 - ((I - J) << 4),
                            ub0 + (J << 4), acc0);

        if (act) {
            const long ob = ((long)(bb * DMODEL + d)) * TLEN + (I << 4) + (hi << 3);
            v4f p0, p1;
            p0[0] = acc0[0] + acc1[0]; p0[1] = acc0[1] + acc1[1];
            p0[2] = acc0[2] + acc1[2]; p0[3] = acc0[3] + acc1[3];
            p1[0] = acc0[4] + acc1[4]; p1[1] = acc0[5] + acc1[5];
            p1[2] = acc0[6] + acc1[6]; p1[3] = acc0[7] + acc1[7];
            __builtin_nontemporal_store(p0, (v4f*)(y + ob));
            __builtin_nontemporal_store(p1, (v4f*)(y + ob + 4));
        }
    }
}

extern "C" void kernel_launch(void* const* d_in, const int* in_sizes, int n_in,
                              void* d_out, int out_size, void* d_ws, size_t ws_size,
                              hipStream_t stream) {
    const float* u     = (const float*)d_in[0];
    const float* Alog  = (const float*)d_in[1];
    const float* Aang  = (const float*)d_in[2];
    const float* Bp    = (const float*)d_in[3];
    const float* Cp    = (const float*)d_in[4];
    const float* Dp    = (const float*)d_in[5];
    const float* logdt = (const float*)d_in[6];
    unsigned short* KRws = (unsigned short*)d_ws;   // 512*2080*2 ~= 2.1 MB
    float* y = (float*)d_out;

    s4d_kgen<<<DMODEL, 256, 0, stream>>>(Alog, Aang, Bp, Cp, Dp, logdt, KRws);
    s4d_conv<<<DMODEL, 256, 0, stream>>>(u, KRws, y);
}
